// Propagator_59562606461145
// MI455X (gfx1250) — matrix operand order repaired and checked by reading, not compiled, not run on hardware
//
#include <hip/hip_runtime.h>
#include <stdint.h>

// ---------------------------------------------------------------------------
// CDNA5 (gfx1250) fused GNN message-passing with bf16 WMMA.
//   edge MLP: [E,512] -> 512 -> 512 -> 128  (x2 directions, shared weights)
//   node MLP: [N,256] -> 256 -> 256 -> 128
// bf16 operands, fp32 accumulation (v_wmma_f32_16x16x32_bf16),
// fp32 atomics for segment_sum.
// ---------------------------------------------------------------------------

typedef __attribute__((ext_vector_type(16))) __bf16 v16bf;
typedef __attribute__((ext_vector_type(8)))  float  v8f;

union FragU { uint4 q[2]; v16bf v; };

__device__ __forceinline__ unsigned short f2bf(float f) {
  unsigned int u = __float_as_uint(f);
  u += 0x7fffu + ((u >> 16) & 1u);          // round-to-nearest-even
  return (unsigned short)(u >> 16);
}
__device__ __forceinline__ float lrelu(float x) { return x > 0.f ? x : 0.01f * x; }

#define WMMA_BF16(A, B, C) \
  __builtin_amdgcn_wmma_f32_16x16x32_bf16(false, (A), false, (B), (short)0, (C), false, false)

// ---------------------------------------------------------------------------
// Edge-MLP 512->512 layer. Per block: 64 rows x 512 cols. 8 waves; wave w
// owns columns [w*64, w*64+64) (4 n-tiles), all 4 m-tiles. K = 512.
// ---------------------------------------------------------------------------
__device__ __forceinline__ void edge_layer512(
    const unsigned short* __restrict__ sSrc, unsigned short* __restrict__ sDst,
    const unsigned short* __restrict__ Wt,   // bf16 [512 (N) x 512 (K)]
    const float* __restrict__ bias, int wave, int lane)
{
  const v8f vz = {0.f, 0.f, 0.f, 0.f, 0.f, 0.f, 0.f, 0.f};
  v8f acc[4][4];
#pragma unroll
  for (int mt = 0; mt < 4; ++mt)
#pragma unroll
    for (int nt = 0; nt < 4; ++nt) acc[mt][nt] = vz;

  const int rowA  = lane & 15;
  const int kSelA = (lane < 16) ? 0 : 8;    // A: lanes 0-15 hold K 0..7/16..23, 16-31 hold 8..15/24..31
  const int kSelB = (lane >> 4) << 3;       // B: lanes 0-15 hold K 0..7/16..23, 16-31 hold 8..15/24..31
  const int colLn = lane & 15;

  for (int kt = 0; kt < 16; ++kt) {
    const int kbase = kt * 32;
    FragU A[4];
#pragma unroll
    for (int mt = 0; mt < 4; ++mt) {
      const uint4* p = (const uint4*)&sSrc[(mt * 16 + rowA) * 512 + kbase + kSelA];
      A[mt].q[0] = p[0];                    // K k0..k0+7
      A[mt].q[1] = p[2];                    // K k0+16..k0+23
    }
    FragU B[4];
#pragma unroll
    for (int nt = 0; nt < 4; ++nt) {
      const uint4* p = (const uint4*)&Wt[(size_t)(wave * 64 + nt * 16 + colLn) * 512 + kbase + kSelB];
      B[nt].q[0] = p[0];
      B[nt].q[1] = p[2];                    // K k0..k0+7 and k0+16..k0+23 per lane
    }
#pragma unroll
    for (int mt = 0; mt < 4; ++mt)
#pragma unroll
      for (int nt = 0; nt < 4; ++nt)
        acc[mt][nt] = WMMA_BF16(A[mt].v, B[nt].v, acc[mt][nt]);
  }

  const int mofs = (lane < 16) ? 0 : 8;     // C/D: vgpr j -> M = j + mofs, N = lane&15
#pragma unroll
  for (int mt = 0; mt < 4; ++mt)
#pragma unroll
    for (int nt = 0; nt < 4; ++nt) {
      const int col = wave * 64 + nt * 16 + colLn;
      const float bv = bias[col];
#pragma unroll
      for (int j = 0; j < 8; ++j) {
        float s = acc[mt][nt][j] + bv;
        sDst[(mt * 16 + mofs + j) * 512 + col] = f2bf(lrelu(s));
      }
    }
}

// ---------------------------------------------------------------------------
// Node-MLP 256->256 layer. 64 rows x 256 cols; wave w owns columns
// [w*32, w*32+32) (2 n-tiles). K = 256.
// ---------------------------------------------------------------------------
__device__ __forceinline__ void node_layer256(
    const unsigned short* __restrict__ sSrc, unsigned short* __restrict__ sDst,
    const unsigned short* __restrict__ Wt,   // bf16 [256 (N) x 256 (K)]
    const float* __restrict__ bias, int wave, int lane)
{
  const v8f vz = {0.f, 0.f, 0.f, 0.f, 0.f, 0.f, 0.f, 0.f};
  v8f acc[4][2];
#pragma unroll
  for (int mt = 0; mt < 4; ++mt)
#pragma unroll
    for (int nt = 0; nt < 2; ++nt) acc[mt][nt] = vz;

  const int rowA  = lane & 15;
  const int kSelA = (lane < 16) ? 0 : 8;
  const int kSelB = (lane >> 4) << 3;
  const int colLn = lane & 15;

  for (int kt = 0; kt < 8; ++kt) {
    const int kbase = kt * 32;
    FragU A[4];
#pragma unroll
    for (int mt = 0; mt < 4; ++mt) {
      const uint4* p = (const uint4*)&sSrc[(mt * 16 + rowA) * 256 + kbase + kSelA];
      A[mt].q[0] = p[0];
      A[mt].q[1] = p[2];
    }
    FragU B[2];
#pragma unroll
    for (int nt = 0; nt < 2; ++nt) {
      const uint4* p = (const uint4*)&Wt[(size_t)(wave * 32 + nt * 16 + colLn) * 256 + kbase + kSelB];
      B[nt].q[0] = p[0];
      B[nt].q[1] = p[2];
    }
#pragma unroll
    for (int mt = 0; mt < 4; ++mt)
#pragma unroll
      for (int nt = 0; nt < 2; ++nt)
        acc[mt][nt] = WMMA_BF16(A[mt].v, B[nt].v, acc[mt][nt]);
  }

  const int mofs = (lane < 16) ? 0 : 8;
#pragma unroll
  for (int mt = 0; mt < 4; ++mt)
#pragma unroll
    for (int nt = 0; nt < 2; ++nt) {
      const int col = wave * 32 + nt * 16 + colLn;
      const float bv = bias[col];
#pragma unroll
      for (int j = 0; j < 8; ++j) {
        float s = acc[mt][nt][j] + bv;
        sDst[(mt * 16 + mofs + j) * 256 + col] = f2bf(lrelu(s));
      }
    }
}

// ---------------------------------------------------------------------------
// Fused edge MLP + masked scatter-add.
// grid.x = edge tiles of 64, grid.y = direction (0: u->, 1: v->).
// ---------------------------------------------------------------------------
__global__ __launch_bounds__(256)
void edge_mlp_scatter(const unsigned short* __restrict__ nv_bf,
                      const unsigned short* __restrict__ nv0_bf,
                      const int* __restrict__ u_idx,
                      const int* __restrict__ v_idx,
                      const int* __restrict__ et,
                      const unsigned short* __restrict__ W1t,
                      const unsigned short* __restrict__ W2t,
                      const unsigned short* __restrict__ W3t,  // [128 x 512]
                      const float* __restrict__ b1,
                      const float* __restrict__ b2,
                      const float* __restrict__ b3,
                      float* __restrict__ agg,
                      int n_edges, int mask_target)
{
  extern __shared__ char smem[];
  unsigned short* sX = (unsigned short*)smem;   // 64 x 512 bf16 (input, then h2)
  unsigned short* sH = sX + 64 * 512;           // 64 x 512 bf16 (h1)
  int*   sIdx  = (int*)(sH + 64 * 512);         // scatter index per row
  float* sMask = (float*)(sIdx + 64);           // mask per row

  const int tid  = threadIdx.x;
  const int wave = tid >> 5;
  const int lane = tid & 31;
  const int e0   = blockIdx.x * 64;
  const int dir  = blockIdx.y;

  // ---- gather: row = [nv[a] | nv[b] | nv0[a] | nv0[b]] (bf16) ----
  {
    int r = tid >> 2, q = tid & 3;
    int e = e0 + r;
    uint4* dst = (uint4*)&sX[r * 512 + q * 128];
    if (e < n_edges) {
      int uu = u_idx[e], vv = v_idx[e];
      int a = dir ? vv : uu;
      int b = dir ? uu : vv;
      const unsigned short* src =
          (q == 0) ? nv_bf  + (size_t)a * 128 :
          (q == 1) ? nv_bf  + (size_t)b * 128 :
          (q == 2) ? nv0_bf + (size_t)a * 128 :
                     nv0_bf + (size_t)b * 128;
      const uint4* s4 = (const uint4*)src;
#pragma unroll
      for (int i = 0; i < 8; ++i) dst[i] = s4[i];
      if (q == 0) { sIdx[r] = a; sMask[r] = (et[e] == mask_target) ? 1.f : 0.f; }
    } else {
      uint4 z = make_uint4(0, 0, 0, 0);
#pragma unroll
      for (int i = 0; i < 8; ++i) dst[i] = z;
      if (q == 0) { sIdx[r] = 0; sMask[r] = 0.f; }
    }
  }
  __syncthreads();

  edge_layer512(sX, sH, W1t, b1, wave, lane);   // h1 = lrelu(X @ W1 + b1)
  __syncthreads();
  edge_layer512(sH, sX, W2t, b2, wave, lane);   // h2 = lrelu(h1 @ W2 + b2)
  __syncthreads();

  // ---- layer 3 (512 -> 128) + masked atomic scatter ----
  {
    const v8f vz = {0.f, 0.f, 0.f, 0.f, 0.f, 0.f, 0.f, 0.f};
    v8f acc[4] = {vz, vz, vz, vz};
    const int rowA  = lane & 15;
    const int kSelA = (lane < 16) ? 0 : 8;
    const int kSelB = (lane >> 4) << 3;
    const int col   = wave * 16 + (lane & 15);  // wave w owns n-tile w

    for (int kt = 0; kt < 16; ++kt) {
      const int kbase = kt * 32;
      FragU B;
      {
        const uint4* p = (const uint4*)&W3t[(size_t)col * 512 + kbase + kSelB];
        B.q[0] = p[0]; B.q[1] = p[2];
      }
#pragma unroll
      for (int mt = 0; mt < 4; ++mt) {
        FragU A;
        const uint4* p = (const uint4*)&sX[(mt * 16 + rowA) * 512 + kbase + kSelA];
        A.q[0] = p[0]; A.q[1] = p[2];
        acc[mt] = WMMA_BF16(A.v, B.v, acc[mt]);
      }
    }
    const float bv = b3[col];
    const int mofs = (lane < 16) ? 0 : 8;
#pragma unroll
    for (int mt = 0; mt < 4; ++mt)
#pragma unroll
      for (int j = 0; j < 8; ++j) {
        int row = mt * 16 + mofs + j;
        int e = e0 + row;
        float m = sMask[row];
        if (e < n_edges && m != 0.f)
          atomicAdd(&agg[(size_t)sIdx[row] * 128 + col], (acc[mt][j] + bv) * m);
      }
  }
}

// ---------------------------------------------------------------------------
// Fused node MLP: X = [bf16(agg) | nv_bf] (64 x 256) -> 256 -> 256 -> 128.
// Writes fp32 result and refreshed bf16 node state (in place, safe per block).
// ---------------------------------------------------------------------------
__global__ __launch_bounds__(256)
void node_mlp(const float* __restrict__ agg,
              unsigned short* __restrict__ nv_bf,       // in/out
              const unsigned short* __restrict__ W1t,   // [256 x 256]
              const unsigned short* __restrict__ W2t,
              const unsigned short* __restrict__ W3t,   // [128 x 256]
              const float* __restrict__ b1,
              const float* __restrict__ b2,
              const float* __restrict__ b3,
              float* __restrict__ nv_out_f32,
              int n_nodes)
{
  extern __shared__ char smem[];
  unsigned short* sX = (unsigned short*)smem;   // 64 x 256
  unsigned short* sH = sX + 64 * 256;           // 64 x 256

  const int tid  = threadIdx.x;
  const int wave = tid >> 5;
  const int lane = tid & 31;
  const int n0   = blockIdx.x * 64;

  // ---- gather ----
  {
    int r = tid >> 2, q = tid & 3;
    int n = n0 + r;
    unsigned short* dst = &sX[r * 256 + q * 64];
    if (n < n_nodes) {
      if (q < 2) {                              // agg fp32 -> bf16
        const float* s = agg + (size_t)n * 128 + q * 64;
#pragma unroll
        for (int i = 0; i < 64; ++i) dst[i] = f2bf(s[i]);
      } else {                                  // nv bf16 copy
        const uint4* s4 = (const uint4*)(nv_bf + (size_t)n * 128 + (q - 2) * 64);
        uint4* d4 = (uint4*)dst;
#pragma unroll
        for (int i = 0; i < 8; ++i) d4[i] = s4[i];
      }
    } else {
      uint4* d4 = (uint4*)dst;
      uint4 z = make_uint4(0, 0, 0, 0);
#pragma unroll
      for (int i = 0; i < 8; ++i) d4[i] = z;
    }
  }
  __syncthreads();

  node_layer256(sX, sH, W1t, b1, wave, lane);
  __syncthreads();
  node_layer256(sH, sX, W2t, b2, wave, lane);
  __syncthreads();

  // ---- layer 3 (256 -> 128), fp32 out + bf16 state refresh ----
  {
    const v8f vz = {0.f, 0.f, 0.f, 0.f, 0.f, 0.f, 0.f, 0.f};
    v8f acc[4] = {vz, vz, vz, vz};
    const int rowA  = lane & 15;
    const int kSelA = (lane < 16) ? 0 : 8;
    const int kSelB = (lane >> 4) << 3;
    const int col   = wave * 16 + (lane & 15);

    for (int kt = 0; kt < 8; ++kt) {
      const int kbase = kt * 32;
      FragU B;
      {
        const uint4* p = (const uint4*)&W3t[(size_t)col * 256 + kbase + kSelB];
        B.q[0] = p[0]; B.q[1] = p[2];
      }
#pragma unroll
      for (int mt = 0; mt < 4; ++mt) {
        FragU A;
        const uint4* p = (const uint4*)&sX[(mt * 16 + rowA) * 256 + kbase + kSelA];
        A.q[0] = p[0]; A.q[1] = p[2];
        acc[mt] = WMMA_BF16(A.v, B.v, acc[mt]);
      }
    }
    const float bv = b3[col];
    const int mofs = (lane < 16) ? 0 : 8;
#pragma unroll
    for (int mt = 0; mt < 4; ++mt)
#pragma unroll
      for (int j = 0; j < 8; ++j) {
        int row = mt * 16 + mofs + j;
        int n = n0 + row;
        if (n < n_nodes) {
          float s = acc[mt][j] + bv;
          nv_out_f32[(size_t)n * 128 + col] = s;
          nv_bf[(size_t)n * 128 + col]      = f2bf(s);
        }
      }
  }
}

// ---------------------------------------------------------------------------
// Prep kernels
// ---------------------------------------------------------------------------
__global__ void k_transpose_bf16(const float* __restrict__ W,
                                 unsigned short* __restrict__ Wt, int K, int N)
{
  int i = blockIdx.x * blockDim.x + threadIdx.x;
  if (i < K * N) {
    int k = i / N, n = i % N;
    Wt[(size_t)n * K + k] = f2bf(W[i]);       // [K,N] fp32 -> [N,K] bf16
  }
}

__global__ void k_cast_bf16(const float* __restrict__ x,
                            unsigned short* __restrict__ y, int n)
{
  int i = blockIdx.x * blockDim.x + threadIdx.x;
  if (i < n) y[i] = f2bf(x[i]);
}

// ---------------------------------------------------------------------------
// Host orchestration
// ---------------------------------------------------------------------------
extern "C" void kernel_launch(void* const* d_in, const int* in_sizes, int n_in,
                              void* d_out, int out_size, void* d_ws, size_t ws_size,
                              hipStream_t stream)
{
  const int NN = 50000, NE = 100000;
  const float* nv_in  = (const float*)d_in[0];
  const float* nv0_in = (const float*)d_in[1];
  const int* u  = (const int*)d_in[2];
  const int* v  = (const int*)d_in[3];
  const int* et = (const int*)d_in[4];

  // params flattened in setup_inputs() dict insertion order:
  // groups f_ef, f_ef2, f_n, f_n2; each list [0],[1]; each dict W1,b1,W2,b2,W3,b3
  auto P = [&](int g, int r, int item) -> const float* {
    return (const float*)d_in[5 + g * 12 + r * 6 + item];
  };

  char* ws = (char*)d_ws;
  auto carve = [&](size_t bytes) -> char* {
    char* p = ws;
    ws += (bytes + 255) & ~(size_t)255;
    return p;
  };
  float*          agg    = (float*)carve((size_t)NN * 128 * 4);
  unsigned short* nv_bf  = (unsigned short*)carve((size_t)NN * 128 * 2);
  unsigned short* nv0_bf = (unsigned short*)carve((size_t)NN * 128 * 2);

  const int eK[3] = {512, 512, 512}, eN[3] = {512, 512, 128};
  const int nK[3] = {256, 256, 256}, nN[3] = {256, 256, 128};
  unsigned short* eW[4][3];
  unsigned short* nW[4][3];
  for (int p = 0; p < 4; ++p) {
    for (int l = 0; l < 3; ++l) eW[p][l] = (unsigned short*)carve((size_t)eK[l] * eN[l] * 2);
    for (int l = 0; l < 3; ++l) nW[p][l] = (unsigned short*)carve((size_t)nK[l] * nN[l] * 2);
  }

  // weight prep: transpose + bf16 conversion (tiny, weights stay L2-resident)
  for (int p = 0; p < 4; ++p) {
    int r = p >> 1, ge = p & 1, gn = 2 + (p & 1);
    for (int l = 0; l < 3; ++l) {
      int tot = eK[l] * eN[l];
      k_transpose_bf16<<<(tot + 255) / 256, 256, 0, stream>>>(P(ge, r, l * 2), eW[p][l], eK[l], eN[l]);
      int totn = nK[l] * nN[l];
      k_transpose_bf16<<<(totn + 255) / 256, 256, 0, stream>>>(P(gn, r, l * 2), nW[p][l], nK[l], nN[l]);
    }
  }
  k_cast_bf16<<<(NN * 128 + 255) / 256, 256, 0, stream>>>(nv_in,  nv_bf,  NN * 128);
  k_cast_bf16<<<(NN * 128 + 255) / 256, 256, 0, stream>>>(nv0_in, nv0_bf, NN * 128);

  const size_t edgeShm = (size_t)2 * 64 * 512 * 2 + 64 * 4 + 64 * 4;  // 128.5 KB (of 320 KB/WGP)
  const size_t nodeShm = (size_t)2 * 64 * 256 * 2;                    // 64 KB

  for (int p = 0; p < 4; ++p) {
    int r = p >> 1, ge = p & 1, gn = 2 + (p & 1);
    hipMemsetAsync(agg, 0, (size_t)NN * 128 * 4, stream);
    dim3 eg((NE + 63) / 64, 2);
    edge_mlp_scatter<<<eg, 256, edgeShm, stream>>>(
        nv_bf, nv0_bf, u, v, et,
        eW[p][0], eW[p][1], eW[p][2],
        P(ge, r, 1), P(ge, r, 3), P(ge, r, 5),
        agg, NE, p & 1);
    node_mlp<<<(NN + 63) / 64, 256, nodeShm, stream>>>(
        agg, nv_bf,
        nW[p][0], nW[p][1], nW[p][2],
        P(gn, r, 1), P(gn, r, 3), P(gn, r, 5),
        (float*)d_out, NN);
  }
}
